// HEARD_69123203662250
// MI455X (gfx1250) — matrix operand order repaired and checked by reading, not compiled, not run on hardware
//
#include <hip/hip_runtime.h>
#include <cstdint>
#include <cstddef>

// ---------------------------------------------------------------------------
// CT-LSTM scan for MI455X (gfx1250, wave32, WMMA bf16 -> f32)
// B=256, T=256, H=512 ; W is [2H, 7H] fp32 ; out h_seq is [T,B,H] fp32
//
// Two paths, selected by ws_size:
//  BIG path (ws >= ~479MB): zx = x @ Wx + b precomputed for ALL timesteps in
//    one massively parallel GEMM (M=65536); the sequential scan then only
//    runs the recurrent half (K=512), halving the dependent-WMMA chain.
//  SMALL path: round-3 fused kernel (K=1024 per step, bias-init accum).
// Both: 7 gate column-chunks per block (no z materialization), LDS
// double-buffered staging, all B fragments in distinct registers.
// ---------------------------------------------------------------------------

typedef __bf16 bf16;
typedef bf16  v16bf __attribute__((ext_vector_type(16)));
typedef bf16  v8bf  __attribute__((ext_vector_type(8)));
typedef bf16  v4bf  __attribute__((ext_vector_type(4)));
typedef float v8f   __attribute__((ext_vector_type(8)));

#define B_  256
#define T_  256
#define H_  512
#define K2  1024   // 2H
#define N7  3584   // 7H

// workspace layout (bytes)
#define WT_BYTES  ((size_t)K2 * N7 * 2)          // bf16 W transposed [N7][K2]
#define HB_BYTES  ((size_t)B_ * H_ * 2)          // bf16 h buffer
#define C_BYTES   ((size_t)B_ * H_ * 4)          // f32 c / cbar
#define ZX_OFF    (WT_BYTES + 2 * HB_BYTES + 2 * C_BYTES)
#define ZX_BYTES  ((size_t)B_ * T_ * N7 * 2)     // bf16 zx [B*T][7H]

#define KC   32     // K per LDS stage (one WMMA K-step)
#define LDA  40     // bf16 row stride in LDS (80B rows -> all frags 16B aligned)

__device__ __forceinline__ float sigf_(float v) {
    return 1.0f / (1.0f + __expf(-v));
}
__device__ __forceinline__ float softplusf_(float v) {
    return (v > 20.0f) ? v : __logf(1.0f + __expf(v));
}

// ---- phase 0a: W [K2][N7] f32 -> Wt [N7][K2] bf16 (transposed, once) ------
__global__ void wt_convert_kernel(const float* __restrict__ W,
                                  bf16* __restrict__ Wt) {
    size_t idx = (size_t)blockIdx.x * blockDim.x + threadIdx.x;
    if (idx >= (size_t)K2 * N7) return;
    int n = (int)(idx % N7);
    int k = (int)(idx / N7);
    Wt[(size_t)n * K2 + k] = (bf16)W[idx];
}

// ---- phase 0b: zero h0 / c0 / cbar0 ---------------------------------------
__global__ void zero_kernel(uint32_t* __restrict__ p, size_t nwords) {
    size_t i = (size_t)blockIdx.x * blockDim.x + threadIdx.x;
    if (i < nwords) p[i] = 0u;
}

// ===========================================================================
// shared LDS staging / compute helpers
// ===========================================================================

// B: 224 rows (7 gates x 32 cols) x 32 k ; 4-elem (8B) chunks, coalesced.
__device__ __forceinline__ void stage_B(
    bf16* __restrict__ Bsb, const bf16* __restrict__ Wt,
    int tid, int n0, int kw)   // kw = k offset within Wt row
{
#pragma unroll
    for (int j = 0; j < 7; ++j) {
        const int cidx = tid + 256 * j;
        const int row  = cidx >> 3;        // 0..223 = g*32 + nc
        const int koff = (cidx & 7) * 4;   // 0..28
        const int g    = row >> 5;
        const int nc   = row & 31;
        const bf16* src = Wt + (size_t)(g * H_ + n0 + nc) * K2 + (size_t)(kw + koff);
        *(v4bf*)(&Bsb[row * LDA + koff]) = *(const v4bf*)src;
    }
}

// A from f32 rows (converted to bf16): rows are [row][H_] with given base row.
__device__ __forceinline__ void stage_A_f32(
    bf16* __restrict__ Asb, const float* __restrict__ rowbase,
    int tid, int k0)
{
    const int arow = tid >> 2;          // 0..63
    const int acol = (tid & 3) * 8;     // 0,8,16,24
    const float* src = rowbase + (size_t)arow * H_ + (k0 + acol);
    const float4 f0 = *(const float4*)(src + 0);
    const float4 f1 = *(const float4*)(src + 4);
    v8bf v;
    v[0] = (bf16)f0.x; v[1] = (bf16)f0.y; v[2] = (bf16)f0.z; v[3] = (bf16)f0.w;
    v[4] = (bf16)f1.x; v[5] = (bf16)f1.y; v[6] = (bf16)f1.z; v[7] = (bf16)f1.w;
    *(v8bf*)(&Asb[arow * LDA + acol]) = v;
}

// A from bf16 rows: rows are [row][H_] with given base row.
__device__ __forceinline__ void stage_A_bf16(
    bf16* __restrict__ Asb, const bf16* __restrict__ rowbase,
    int tid, int k0)
{
    const int arow = tid >> 2;
    const int acol = (tid & 3) * 8;
    const bf16* src = rowbase + (size_t)arow * H_ + (k0 + acol);
    *(v8bf*)(&Asb[arow * LDA + acol]) = *(const v8bf*)src;
}

// one K-step of 7 WMMAs; all B fragments loaded into distinct registers first
__device__ __forceinline__ void compute_chunk(
    const bf16* __restrict__ Asb, const bf16* __restrict__ Bsb,
    int abase, int bbase, v8f acc[7])
{
    const v8bf a_lo = *(const v8bf*)(&Asb[abase]);
    const v8bf a_hi = *(const v8bf*)(&Asb[abase + 16]);
    const v16bf afrag = __builtin_shufflevector(
        a_lo, a_hi, 0, 1, 2, 3, 4, 5, 6, 7, 8, 9, 10, 11, 12, 13, 14, 15);

    v16bf bfrag[7];
#pragma unroll
    for (int g = 0; g < 7; ++g) {
        const bf16* bp  = &Bsb[g * 32 * LDA + bbase];
        const v8bf b_lo = *(const v8bf*)(bp);
        const v8bf b_hi = *(const v8bf*)(bp + 16);
        bfrag[g] = __builtin_shufflevector(
            b_lo, b_hi, 0, 1, 2, 3, 4, 5, 6, 7, 8, 9, 10, 11, 12, 13, 14, 15);
    }
#pragma unroll
    for (int g = 0; g < 7; ++g) {
        acc[g] = __builtin_amdgcn_wmma_f32_16x16x32_bf16(
            false, afrag, false, bfrag[g], (short)0, acc[g], false, false);
    }
}

// common tile geometry
struct TileIdx {
    int tid, lane, wave, wm, wn, mloc, hi, abase, bbase;
};
__device__ __forceinline__ TileIdx tile_idx() {
    TileIdx s;
    s.tid  = threadIdx.x;
    s.lane = s.tid & 31;
    s.wave = s.tid >> 5;
    s.wm   = s.wave & 3;
    s.wn   = s.wave >> 2;
    s.mloc = s.lane & 15;
    s.hi   = s.lane >> 4;
    s.abase = (s.wm * 16 + s.mloc) * LDA + s.hi * 8;
    s.bbase = (s.wn * 16 + s.mloc) * LDA + s.hi * 8;
    return s;
}

// ===========================================================================
// BIG path kernel 1: zx[m][7H] = x[m][:H] @ Wx + b   (m = b*T + t, parallel)
// grid = (65536/64, 512/32) = (1024, 16)
// ===========================================================================
__global__ __launch_bounds__(256)
void zx_gemm_kernel(const float* __restrict__ x,
                    const bf16*  __restrict__ Wt,
                    const float* __restrict__ bias,
                    bf16*        __restrict__ zx) {
    __shared__ __align__(16) bf16 As[2][64  * LDA];
    __shared__ __align__(16) bf16 Bs[2][224 * LDA];

    const TileIdx s = tile_idx();
    const int m0 = blockIdx.x * 64;
    const int n0 = blockIdx.y * 32;
    const int ncol = n0 + s.wn * 16 + s.mloc;
    const float* rowbase = x + (size_t)m0 * H_;

    v8f acc[7];
#pragma unroll
    for (int g = 0; g < 7; ++g) {
        const float bb = bias[g * H_ + ncol];
#pragma unroll
        for (int i = 0; i < 8; ++i) acc[g][i] = bb;
    }

    constexpr int NC = H_ / KC;   // 16
    stage_A_f32(As[0], rowbase, s.tid, 0);
    stage_B(Bs[0], Wt, s.tid, n0, 0);
#pragma unroll 2
    for (int i = 0; i < NC - 1; ++i) {
        __syncthreads();
        stage_A_f32(As[(i + 1) & 1], rowbase, s.tid, (i + 1) * KC);
        stage_B(Bs[(i + 1) & 1], Wt, s.tid, n0, (i + 1) * KC);
        compute_chunk(As[i & 1], Bs[i & 1], s.abase, s.bbase, acc);
    }
    __syncthreads();
    compute_chunk(As[(NC - 1) & 1], Bs[(NC - 1) & 1], s.abase, s.bbase, acc);

#pragma unroll
    for (int i = 0; i < 8; ++i) {
        const int m = m0 + s.wm * 16 + i + 8 * s.hi;
#pragma unroll
        for (int g = 0; g < 7; ++g)
            zx[(size_t)m * N7 + g * H_ + ncol] = (bf16)acc[g][i];
    }
}

// ===========================================================================
// gate/decay epilogue shared by both step kernels
// ===========================================================================
__device__ __forceinline__ void ctlstm_epilogue(
    const v8f acc[7], const float* __restrict__ dt,
    bf16* __restrict__ hout, float* __restrict__ c, float* __restrict__ cbar,
    float* __restrict__ out, int t, int bm0, const TileIdx& s, int ncol)
{
#pragma unroll
    for (int i = 0; i < 8; ++i) {
        const int   b_idx = bm0 + s.wm * 16 + i + 8 * s.hi;
        const float gi  = sigf_(acc[0][i]);
        const float gf  = sigf_(acc[1][i]);
        const float go  = sigf_(acc[2][i]);
        const float gz  = tanhf(acc[3][i]);
        const float gib = sigf_(acc[4][i]);
        const float gfb = sigf_(acc[5][i]);
        const float gd  = softplusf_(acc[6][i]);

        const size_t sidx   = (size_t)b_idx * H_ + ncol;
        const float  c_old  = c[sidx];
        const float  cb_old = cbar[sidx];
        const float  dtv    = dt[(size_t)b_idx * T_ + t];

        const float c_new  = gf * c_old + gi * gz;
        const float cb_new = gfb * cb_old + gib * gz;
        const float c_dec  = cb_new + (c_new - cb_new) * __expf(-gd * dtv);
        const float h_dec  = go * tanhf(c_dec);

        c[sidx]    = c_dec;      // carry is decayed cell
        cbar[sidx] = cb_new;
        out[((size_t)t * B_ + b_idx) * H_ + ncol] = h_dec;
        hout[sidx] = (bf16)h_dec;
    }
}

// ===========================================================================
// BIG path kernel 2: recurrent-only step (K=512), accum init from zx
// ===========================================================================
__global__ __launch_bounds__(256)
void ctlstm_step_h_kernel(const float* __restrict__ dt,
                          const bf16*  __restrict__ Wt,
                          const bf16*  __restrict__ zx,
                          const bf16*  __restrict__ hin,
                          bf16*        __restrict__ hout,
                          float*       __restrict__ c,
                          float*       __restrict__ cbar,
                          float*       __restrict__ out,
                          int t) {
    __shared__ __align__(16) bf16 As[2][64  * LDA];
    __shared__ __align__(16) bf16 Bs[2][224 * LDA];

    const TileIdx s = tile_idx();
    const int bm0 = blockIdx.x * 64;
    const int n0  = blockIdx.y * 32;
    const int ncol = n0 + s.wn * 16 + s.mloc;
    const bf16* rowbase = hin + (size_t)bm0 * H_;

    // accum init = zx (x-part + bias, precomputed) for each (row, gate)
    v8f acc[7];
#pragma unroll
    for (int i = 0; i < 8; ++i) {
        const int m = (bm0 + s.wm * 16 + i + 8 * s.hi) * T_ + t;  // zx row b*T+t
#pragma unroll
        for (int g = 0; g < 7; ++g)
            acc[g][i] = (float)zx[(size_t)m * N7 + g * H_ + ncol];
    }

    constexpr int NC = H_ / KC;   // 16 (recurrent half only)
    stage_A_bf16(As[0], rowbase, s.tid, 0);
    stage_B(Bs[0], Wt, s.tid, n0, H_ + 0);
#pragma unroll 2
    for (int i = 0; i < NC - 1; ++i) {
        __syncthreads();
        stage_A_bf16(As[(i + 1) & 1], rowbase, s.tid, (i + 1) * KC);
        stage_B(Bs[(i + 1) & 1], Wt, s.tid, n0, H_ + (i + 1) * KC);
        compute_chunk(As[i & 1], Bs[i & 1], s.abase, s.bbase, acc);
    }
    __syncthreads();
    compute_chunk(As[(NC - 1) & 1], Bs[(NC - 1) & 1], s.abase, s.bbase, acc);

    ctlstm_epilogue(acc, dt, hout, c, cbar, out, t, bm0, s, ncol);
}

// ===========================================================================
// SMALL path: fully fused step (K=1024, bias init) — round-3 kernel
// ===========================================================================
__global__ __launch_bounds__(256)
void ctlstm_step_full_kernel(const float* __restrict__ x,
                             const float* __restrict__ dt,
                             const bf16*  __restrict__ Wt,
                             const float* __restrict__ bias,
                             const bf16*  __restrict__ hin,
                             bf16*        __restrict__ hout,
                             float*       __restrict__ c,
                             float*       __restrict__ cbar,
                             float*       __restrict__ out,
                             int t) {
    __shared__ __align__(16) bf16 As[2][64  * LDA];
    __shared__ __align__(16) bf16 Bs[2][224 * LDA];

    const TileIdx s = tile_idx();
    const int bm0 = blockIdx.x * 64;
    const int n0  = blockIdx.y * 32;
    const int ncol = n0 + s.wn * 16 + s.mloc;
    // x rows for this t: x[b][t][:] ; stride between b rows is T_*H_
    const float* xrow = x + ((size_t)bm0 * T_ + t) * H_;
    const bf16*  hrow = hin + (size_t)bm0 * H_;

    v8f acc[7];
#pragma unroll
    for (int g = 0; g < 7; ++g) {
        const float bb = bias[g * H_ + ncol];
#pragma unroll
        for (int i = 0; i < 8; ++i) acc[g][i] = bb;
    }

    constexpr int NC = K2 / KC;   // 32

    // A staging for the full kernel: x rows have stride T_*H_, h rows H_.
    // stage inline (x rows are not contiguous in m, so can't use stage_A_f32)
    const int arow = s.tid >> 2;
    const int acol = (s.tid & 3) * 8;

#define STAGE_FULL(buf, k0_)                                                     \
    do {                                                                         \
        const int k0v = (k0_);                                                   \
        if (k0v < H_) {                                                          \
            const float* srcp = xrow + (size_t)arow * (T_ * H_) + (k0v + acol);  \
            const float4 f0 = *(const float4*)(srcp + 0);                        \
            const float4 f1 = *(const float4*)(srcp + 4);                        \
            v8bf v;                                                              \
            v[0]=(bf16)f0.x; v[1]=(bf16)f0.y; v[2]=(bf16)f0.z; v[3]=(bf16)f0.w;  \
            v[4]=(bf16)f1.x; v[5]=(bf16)f1.y; v[6]=(bf16)f1.z; v[7]=(bf16)f1.w;  \
            *(v8bf*)(&(buf)[arow * LDA + acol]) = v;                             \
        } else {                                                                 \
            const bf16* srcp = hrow + (size_t)arow * H_ + (k0v - H_ + acol);     \
            *(v8bf*)(&(buf)[arow * LDA + acol]) = *(const v8bf*)srcp;            \
        }                                                                        \
    } while (0)

    STAGE_FULL(As[0], 0);
    stage_B(Bs[0], Wt, s.tid, n0, 0);
#pragma unroll 2
    for (int i = 0; i < NC - 1; ++i) {
        __syncthreads();
        STAGE_FULL(As[(i + 1) & 1], (i + 1) * KC);
        stage_B(Bs[(i + 1) & 1], Wt, s.tid, n0, (i + 1) * KC);
        compute_chunk(As[i & 1], Bs[i & 1], s.abase, s.bbase, acc);
    }
    __syncthreads();
    compute_chunk(As[(NC - 1) & 1], Bs[(NC - 1) & 1], s.abase, s.bbase, acc);
#undef STAGE_FULL

    ctlstm_epilogue(acc, dt, hout, c, cbar, out, t, bm0, s, ncol);
}

// ---------------------------------------------------------------------------
extern "C" void kernel_launch(void* const* d_in, const int* in_sizes, int n_in,
                              void* d_out, int out_size, void* d_ws, size_t ws_size,
                              hipStream_t stream) {
    (void)in_sizes; (void)n_in; (void)out_size;

    const float* x  = (const float*)d_in[0];   // [B,T,H]
    const float* dt = (const float*)d_in[1];   // [B,T]
    const float* W  = (const float*)d_in[2];   // [2H,7H]
    const float* b  = (const float*)d_in[3];   // [7H]
    float* out = (float*)d_out;                // [T,B,H]

    char* ws = (char*)d_ws;
    bf16*  Wt   = (bf16*)(ws);
    bf16*  hA   = (bf16*)(ws + WT_BYTES);
    bf16*  hB   = (bf16*)(ws + WT_BYTES + HB_BYTES);
    float* c    = (float*)(ws + WT_BYTES + 2 * HB_BYTES);
    float* cbar = (float*)(ws + WT_BYTES + 2 * HB_BYTES + C_BYTES);
    bf16*  zx   = (bf16*)(ws + ZX_OFF);

    // phase 0a: transpose + convert W to bf16 [N7][K2]
    {
        const size_t n = (size_t)K2 * N7;
        wt_convert_kernel<<<(unsigned)((n + 255) / 256), 256, 0, stream>>>(W, Wt);
    }
    // phase 0b: zero hA, hB, c, cbar (contiguous span after Wt)
    {
        const size_t nwords = (2 * HB_BYTES + 2 * C_BYTES) / 4;
        zero_kernel<<<(unsigned)((nwords + 255) / 256), 256, 0, stream>>>(
            (uint32_t*)(ws + WT_BYTES), nwords);
    }

    dim3 grid(B_ / 64, H_ / 32);
    if (ws_size >= ZX_OFF + ZX_BYTES) {
        // BIG path: hoist x@Wx+b out of the scan (parallel), scan runs K=512
        zx_gemm_kernel<<<dim3((B_ * T_) / 64, H_ / 32), 256, 0, stream>>>(
            x, Wt, b, zx);
        for (int t = 0; t < T_; ++t) {
            const bf16* hin  = (t & 1) ? hB : hA;
            bf16*       hout = (t & 1) ? hA : hB;
            ctlstm_step_h_kernel<<<grid, 256, 0, stream>>>(
                dt, Wt, zx, hin, hout, c, cbar, out, t);
        }
    } else {
        // SMALL path: fully fused step, K=1024
        for (int t = 0; t < T_; ++t) {
            const bf16* hin  = (t & 1) ? hB : hA;
            bf16*       hout = (t & 1) ? hA : hB;
            ctlstm_step_full_kernel<<<grid, 256, 0, stream>>>(
                x, dt, Wt, b, hin, hout, c, cbar, out, t);
        }
    }
}
